// SharedAttention_45320494907960
// MI455X (gfx1250) — matrix operand order repaired and checked by reading, not compiled, not run on hardware
//
#include <hip/hip_runtime.h>

// ---------------------------------------------------------------------------
// SharedAttention for MI455X (gfx1250, wave32, WMMA).
// All large contractions use v_wmma_f32_16x16x32_bf16 (f32 accumulate).
// bf16-resident operands are staged into LDS with GLOBAL_LOAD_ASYNC_TO_LDS
// (ASYNCcnt-tracked DMA, no VGPR round trip) when the builtin is available.
// ---------------------------------------------------------------------------

typedef __bf16 bf16;
typedef __attribute__((ext_vector_type(8)))  bf16  v8bf;
typedef __attribute__((ext_vector_type(16))) bf16  v16bf;
typedef __attribute__((ext_vector_type(8)))  float v8f;

typedef __attribute__((__vector_size__(4 * sizeof(int)))) int i32x4;
typedef i32x4 __attribute__((address_space(1)))* gas_b128_t;  // global AS pointer
typedef i32x4 __attribute__((address_space(3)))* las_b128_t;  // LDS AS pointer

static constexpr int BB = 32;    // batch
static constexpr int DD = 1024;  // d
static constexpr int TT = 1024;  // t
static constexpr int HH = 2048;  // hidden
static constexpr int BCH = 8;    // batch chunk for attention middle

#define LDSK 40  // 32 K-elements + 8 bf16 pad per LDS tile row (80 bytes)

#if defined(__gfx1250__) && __has_builtin(__builtin_amdgcn_global_load_async_to_lds_b128)
#define ASYNC_LDS 1
#else
#define ASYNC_LDS 0
#endif

// ---- LDS staging: 128 rows x 32 K of bf16, 256 threads (2 segs of 16/row) --
template<bool F32SRC>
__device__ __forceinline__ void stage_tile(bf16* __restrict__ lds,
                                           const void* __restrict__ gbase,
                                           int ld, int row0, int k0, int tid) {
  const int r   = tid >> 1;
  const int seg = (tid & 1) << 4;
  const size_t off = (size_t)(row0 + r) * (size_t)ld + (size_t)(k0 + seg);
  v8bf o0, o1;
  if constexpr (F32SRC) {
    const float4* s4 = (const float4*)((const float*)gbase + off);
    float4 f0 = s4[0], f1 = s4[1], f2 = s4[2], f3 = s4[3];
    o0[0]=(bf16)f0.x; o0[1]=(bf16)f0.y; o0[2]=(bf16)f0.z; o0[3]=(bf16)f0.w;
    o0[4]=(bf16)f1.x; o0[5]=(bf16)f1.y; o0[6]=(bf16)f1.z; o0[7]=(bf16)f1.w;
    o1[0]=(bf16)f2.x; o1[1]=(bf16)f2.y; o1[2]=(bf16)f2.z; o1[3]=(bf16)f2.w;
    o1[4]=(bf16)f3.x; o1[5]=(bf16)f3.y; o1[6]=(bf16)f3.z; o1[7]=(bf16)f3.w;
  } else {
    const bf16* s = (const bf16*)gbase + off;
    o0 = *(const v8bf*)s;
    o1 = *(const v8bf*)(s + 8);
  }
  bf16* d = lds + (size_t)r * LDSK + seg;
  *(v8bf*)d       = o0;
  *(v8bf*)(d + 8) = o1;
}

// Async DMA staging for bf16 sources: global -> LDS without touching VGPRs.
__device__ __forceinline__ void stage_tile_async(bf16* __restrict__ lds,
                                                 const void* __restrict__ gbase,
                                                 int ld, int row0, int k0, int tid) {
#if ASYNC_LDS
  const int r   = tid >> 1;
  const int seg = (tid & 1) << 4;
  const bf16* s = (const bf16*)gbase + (size_t)(row0 + r) * (size_t)ld + (size_t)(k0 + seg);
  bf16* d = lds + (size_t)r * LDSK + seg;
  __builtin_amdgcn_global_load_async_to_lds_b128((gas_b128_t)s,       (las_b128_t)d,       0, 0);
  __builtin_amdgcn_global_load_async_to_lds_b128((gas_b128_t)(s + 8), (las_b128_t)(d + 8), 0, 0);
#else
  stage_tile<false>(lds, gbase, ld, row0, k0, tid);
#endif
}

__device__ __forceinline__ void wait_async() {
#if ASYNC_LDS
  asm volatile("s_wait_asynccnt 0x0" ::: "memory");
#endif
}

// ---- fragment loads per CDNA5 WMMA VGPR layouts (wave32) --------------------
// A 16x32 bf16: lane L -> row M=L%16, K runs [8*(L/16), +8) and [+16, +8)
__device__ __forceinline__ v16bf frag_a(const bf16* tile, int lane) {
  const int m  = lane & 15;
  const int k0 = (lane >> 4) << 3;
  const bf16* p = tile + (size_t)m * LDSK + k0;
  v8bf lo = *(const v8bf*)p;
  v8bf hi = *(const v8bf*)(p + 16);
  return __builtin_shufflevector(lo, hi, 0,1,2,3,4,5,6,7,8,9,10,11,12,13,14,15);
}
// B 32x16 bf16 (stored as Bt: N rows x K): lane L -> col N=L%16, K runs [8*(L/16), +8) and [+16, +8)
__device__ __forceinline__ v16bf frag_b(const bf16* tile, int lane) {
  const int n  = lane & 15;
  const int k0 = (lane >> 4) << 3;
  const bf16* p = tile + (size_t)n * LDSK + k0;
  v8bf lo = *(const v8bf*)p;
  v8bf hi = *(const v8bf*)(p + 16);
  return __builtin_shufflevector(lo, hi, 0,1,2,3,4,5,6,7,8,9,10,11,12,13,14,15);
}

// ---------------------------------------------------------------------------
// Tiled GEMM: C[m,n] = sum_k A[m,k] * Bt[n,k]   (both operands K-contiguous)
// Block tile 128x128, K-step 32, 8 waves (4Mx2N), wave tile 32x64.
// A_ASY/B_ASY: stage that operand with async global->LDS DMA (bf16 only).
// Epilogue: optional bias over M or N, relu, strided f32 store,
//           transposed bf16 store CT[n*ldCT + m] (K-contig feed for next GEMM).
// blockIdx.z batches via element strides sA/sB/sC/sCT.
// ---------------------------------------------------------------------------
template<bool A_F32, bool B_F32, bool A_ASY, bool B_ASY,
         bool BIAS_M, bool BIAS_N, bool RELU, bool ST_F32, bool ST_BF16T>
__global__ __launch_bounds__(256) void gemm_wmma_128x128(
    const void* __restrict__ A,  int ldA, long long sA,
    const void* __restrict__ Bt, int ldB, long long sB,
    const float* __restrict__ biasM, const float* __restrict__ biasN,
    float* __restrict__ C,  int ldC,  long long sC,
    bf16*  __restrict__ CT, int ldCT, long long sCT,
    int K) {
  static_assert(!(A_F32 && A_ASY) && !(B_F32 && B_ASY), "async staging is bf16-only");
  __shared__ bf16 As[2][128 * LDSK];
  __shared__ bf16 Bs[2][128 * LDSK];

  const int tid  = threadIdx.x;
  const int lane = tid & 31;
  const int wave = tid >> 5;
  const int wm   = (wave >> 1) * 32;  // wave M offset in block tile
  const int wn   = (wave & 1) * 64;   // wave N offset in block tile
  const int m0   = blockIdx.y * 128;
  const int n0   = blockIdx.x * 128;
  const int z    = blockIdx.z;

  const void* Ab = A_F32 ? (const void*)((const float*)A + (size_t)z * sA)
                         : (const void*)((const bf16*)A  + (size_t)z * sA);
  const void* Bb = B_F32 ? (const void*)((const float*)Bt + (size_t)z * sB)
                         : (const void*)((const bf16*)Bt  + (size_t)z * sB);

  v8f acc[2][4];
  #pragma unroll
  for (int i = 0; i < 2; ++i)
    #pragma unroll
    for (int j = 0; j < 4; ++j) acc[i][j] = {};

  const int nk = K >> 5;
  if constexpr (A_ASY) stage_tile_async(As[0], Ab, ldA, m0, 0, tid);
  else                 stage_tile<A_F32>(As[0], Ab, ldA, m0, 0, tid);
  if constexpr (B_ASY) stage_tile_async(Bs[0], Bb, ldB, n0, 0, tid);
  else                 stage_tile<B_F32>(Bs[0], Bb, ldB, n0, 0, tid);

  int buf = 0;
  for (int kt = 0; kt < nk; ++kt) {
    if constexpr (A_ASY || B_ASY) wait_async();  // current buf DMA complete
    __syncthreads();
    if (kt + 1 < nk) {
      const int knext = (kt + 1) << 5;
      if constexpr (A_ASY) stage_tile_async(As[buf ^ 1], Ab, ldA, m0, knext, tid);
      else                 stage_tile<A_F32>(As[buf ^ 1], Ab, ldA, m0, knext, tid);
      if constexpr (B_ASY) stage_tile_async(Bs[buf ^ 1], Bb, ldB, n0, knext, tid);
      else                 stage_tile<B_F32>(Bs[buf ^ 1], Bb, ldB, n0, knext, tid);
    }
    v16bf af[2], bfr[4];
    #pragma unroll
    for (int sm = 0; sm < 2; ++sm)
      af[sm] = frag_a(&As[buf][(size_t)(wm + sm * 16) * LDSK], lane);
    #pragma unroll
    for (int sn = 0; sn < 4; ++sn)
      bfr[sn] = frag_b(&Bs[buf][(size_t)(wn + sn * 16) * LDSK], lane);
    #pragma unroll
    for (int sm = 0; sm < 2; ++sm)
      #pragma unroll
      for (int sn = 0; sn < 4; ++sn)
        acc[sm][sn] = __builtin_amdgcn_wmma_f32_16x16x32_bf16(
            false, af[sm], false, bfr[sn], (short)0, acc[sm][sn], false, false);
    buf ^= 1;
  }

  // Epilogue. C/D layout: lane L, reg r -> M = r + 8*(L/16), N = L%16.
  float* Cb = nullptr;
  bf16*  CTb = nullptr;
  if constexpr (ST_F32)   Cb  = C  + (size_t)z * sC;
  if constexpr (ST_BF16T) CTb = CT + (size_t)z * sCT;
  #pragma unroll
  for (int sm = 0; sm < 2; ++sm) {
    #pragma unroll
    for (int sn = 0; sn < 4; ++sn) {
      const int mg = m0 + wm + sm * 16 + ((lane >> 4) << 3);
      const int ng = n0 + wn + sn * 16 + (lane & 15);
      float bn = 0.0f;
      if constexpr (BIAS_N) bn = biasN[ng];
      v8f c = acc[sm][sn];
      float vals[8];
      #pragma unroll
      for (int r = 0; r < 8; ++r) {
        float v = c[r] + bn;
        if constexpr (BIAS_M) v += biasM[mg + r];
        if constexpr (RELU)   v = fmaxf(v, 0.0f);
        vals[r] = v;
      }
      if constexpr (ST_F32) {
        #pragma unroll
        for (int r = 0; r < 8; ++r) Cb[(size_t)(mg + r) * ldC + ng] = vals[r];
      }
      if constexpr (ST_BF16T) {
        v8bf o;
        #pragma unroll
        for (int r = 0; r < 8; ++r) o[r] = (bf16)vals[r];
        *(v8bf*)(CTb + (size_t)ng * ldCT + mg) = o;  // 8 consecutive M, 16B store
      }
    }
  }
}

// ---------------------------------------------------------------------------
// Softmax of exp(qk with diagonal scaled by 1 - 1/sqrt(T)); bf16 row-major out.
// One 256-thread block per (row i, local batch). T = 1024 -> 4 elems/thread.
// ---------------------------------------------------------------------------
__device__ __forceinline__ float wave_max(float v) {
  #pragma unroll
  for (int o = 16; o; o >>= 1) v = fmaxf(v, __shfl_xor(v, o, 32));
  return v;
}
__device__ __forceinline__ float wave_sum(float v) {
  #pragma unroll
  for (int o = 16; o; o >>= 1) v += __shfl_xor(v, o, 32);
  return v;
}

__global__ __launch_bounds__(256) void softmax_kernel(
    const float* __restrict__ qk, bf16* __restrict__ sc) {
  const int i = blockIdx.x;
  const int b = blockIdx.y;
  const float* row = qk + ((size_t)b * TT + i) * TT;
  bf16* orow       = sc + ((size_t)b * TT + i) * TT;
  const int tid = threadIdx.x, lane = tid & 31, wave = tid >> 5;
  __shared__ float red[8];

  float l[4];
  float mx = -3.4e38f;
  #pragma unroll
  for (int t = 0; t < 4; ++t) {
    const int j = tid + t * 256;
    float z = row[j];
    if (j == i) z *= 0.96875f;  // 1 - 1/sqrt(1024)
    l[t] = __expf(z);
    mx = fmaxf(mx, l[t]);
  }
  mx = wave_max(mx);
  if (lane == 0) red[wave] = mx;
  __syncthreads();
  if (tid == 0) {
    float t = red[0];
    #pragma unroll
    for (int w = 1; w < 8; ++w) t = fmaxf(t, red[w]);
    red[0] = t;
  }
  __syncthreads();
  mx = red[0];
  __syncthreads();

  float e[4], s = 0.0f;
  #pragma unroll
  for (int t = 0; t < 4; ++t) { e[t] = __expf(l[t] - mx); s += e[t]; }
  s = wave_sum(s);
  if (lane == 0) red[wave] = s;
  __syncthreads();
  if (tid == 0) {
    float t = 0.0f;
    #pragma unroll
    for (int w = 0; w < 8; ++w) t += red[w];
    red[0] = t;
  }
  __syncthreads();
  const float inv = 1.0f / red[0];
  #pragma unroll
  for (int t = 0; t < 4; ++t) orow[tid + t * 256] = (bf16)(e[t] * inv);
}

// ---------------------------------------------------------------------------
// Ex-path: Tq==1 makes logits collapse to 1 -> uniform softmax -> mean of
// value[b,d,25:1023] (998 elems). One wave per output element everywhere.
// ---------------------------------------------------------------------------
__global__ __launch_bounds__(256) void mean_kernel(
    const float* __restrict__ value, float* __restrict__ xex) {
  const int gw = (blockIdx.x * 256 + threadIdx.x) >> 5;  // (b*D + d)
  const int lane = threadIdx.x & 31;
  const float* src = value + (size_t)gw * TT;
  float s = 0.0f;
  for (int j = 25 + lane; j < 1023; j += 32) s += src[j];
  s = wave_sum(s);
  if (lane == 0) xex[gw] = s * (1.0f / 998.0f);
}

__global__ __launch_bounds__(256) void exmlp1_kernel(
    const float* __restrict__ xex, const float* __restrict__ W1,
    const float* __restrict__ b1, float* __restrict__ hex) {
  const int gw = (blockIdx.x * 256 + threadIdx.x) >> 5;  // (b*H + h)
  const int lane = threadIdx.x & 31;
  const int b = gw >> 11, h = gw & (HH - 1);
  const float* x = xex + (size_t)b * DD;
  const float* w = W1 + (size_t)h * DD;
  float s = 0.0f;
  for (int d = lane; d < DD; d += 32) s += x[d] * w[d];
  s = wave_sum(s);
  if (lane == 0) hex[gw] = fmaxf(s + b1[h], 0.0f);
}

__global__ __launch_bounds__(256) void exmlp2_kernel(
    const float* __restrict__ hex, const float* __restrict__ W2,
    const float* __restrict__ b2, float* __restrict__ rep) {
  const int gw = (blockIdx.x * 256 + threadIdx.x) >> 5;  // (b*D + d)
  const int lane = threadIdx.x & 31;
  const int b = gw >> 10, d = gw & (DD - 1);
  const float* h = hex + (size_t)b * HH;
  const float* w = W2 + (size_t)d * HH;
  float s = 0.0f;
  for (int k = lane; k < HH; k += 32) s += h[k] * w[k];
  s = wave_sum(s);
  if (lane == 0)
    rep[(size_t)b * DD * (TT + 1) + (size_t)d * (TT + 1) + TT] = s + b2[d];
}

// ---------------------------------------------------------------------------
extern "C" void kernel_launch(void* const* d_in, const int* in_sizes, int n_in,
                              void* d_out, int out_size, void* d_ws, size_t ws_size,
                              hipStream_t stream) {
  (void)in_sizes; (void)n_in; (void)out_size; (void)ws_size;
  const float* pattern = (const float*)d_in[0];
  const float* value   = (const float*)d_in[1];
  const float* Wq = (const float*)d_in[2];
  const float* bq = (const float*)d_in[3];
  const float* Wk = (const float*)d_in[4];
  const float* bk = (const float*)d_in[5];
  const float* W1 = (const float*)d_in[6];
  const float* b1 = (const float*)d_in[7];
  const float* W2 = (const float*)d_in[8];
  const float* b2 = (const float*)d_in[9];

  float* out  = (float*)d_out;
  float* rep  = out;                                    // (B, D, T+1)
  float* qout = out + (size_t)BB * DD * (TT + 1);       // (B, D, T)
  float* kout = qout + (size_t)BB * DD * TT;            // (B, D, T)

  // Workspace carve-up (~224 MB)
  char* w = (char*)d_ws;
  bf16* qT  = (bf16*)w;  w += (size_t)BB * TT * DD * sizeof(bf16);   // q transposed (b,t,d)
  bf16* kT  = (bf16*)w;  w += (size_t)BB * TT * DD * sizeof(bf16);
  float* qk = (float*)w; w += (size_t)BCH * TT * TT * sizeof(float); // chunk scores pre-softmax
  bf16* sc  = (bf16*)w;  w += (size_t)BCH * TT * TT * sizeof(bf16);  // softmax scores (row-major == Bt)
  bf16* xT  = (bf16*)w;  w += (size_t)BCH * TT * DD * sizeof(bf16);  // attn out transposed (t,d)
  bf16* hT  = (bf16*)w;  w += (size_t)BCH * TT * HH * sizeof(bf16);  // relu hidden transposed (t,h)
  float* xex = (float*)w; w += (size_t)BB * DD * sizeof(float);
  float* hex = (float*)w; w += (size_t)BB * HH * sizeof(float);

  const long long sBD = (long long)DD * TT;   // per-batch (D,T) stride
  const long long sTT = (long long)TT * TT;
  const long long sTH = (long long)TT * HH;

  // G1: q = pattern @ Wq^T + bq  -> qout (f32) and qT (bf16, (t,d))
  gemm_wmma_128x128<true, true, false, false, false, true, false, true, true>
      <<<dim3(TT / 128, DD / 128, BB), 256, 0, stream>>>(
          pattern, TT, sBD, Wq, TT, 0, nullptr, bq,
          qout, TT, sBD, qT, DD, sBD, TT);
  gemm_wmma_128x128<true, true, false, false, false, true, false, true, true>
      <<<dim3(TT / 128, DD / 128, BB), 256, 0, stream>>>(
          pattern, TT, sBD, Wk, TT, 0, nullptr, bk,
          kout, TT, sBD, kT, DD, sBD, TT);

  for (int b0 = 0; b0 < BB; b0 += BCH) {
    // G2: qk[i,j] = sum_d qT[i,d] * kT[j,d]   (both operands async-DMA bf16)
    gemm_wmma_128x128<false, false, true, true, false, false, false, true, false>
        <<<dim3(TT / 128, TT / 128, BCH), 256, 0, stream>>>(
            qT + (size_t)b0 * sBD, DD, sBD, kT + (size_t)b0 * sBD, DD, sBD,
            nullptr, nullptr, qk, TT, sTT, nullptr, 0, 0, DD);
    // softmax(exp(qk diag-adjusted)) -> sc (bf16)
    softmax_kernel<<<dim3(TT, BCH), 256, 0, stream>>>(qk, sc);
    // G3: attn[d,i] = sum_j value[d,j] * scores[i,j] -> xT (bf16, (i,d))
    gemm_wmma_128x128<true, false, false, true, false, false, false, false, true>
        <<<dim3(TT / 128, DD / 128, BCH), 256, 0, stream>>>(
            value + (size_t)b0 * sBD, TT, sBD, sc, TT, sTT,
            nullptr, nullptr, nullptr, 0, 0, xT, DD, sBD, TT);
    // G4: h[h,t] = relu(sum_d W1[h,d] * xT[t,d] + b1[h]) -> hT (bf16, (t,h))
    gemm_wmma_128x128<true, false, false, true, true, false, true, false, true>
        <<<dim3(TT / 128, HH / 128, BCH), 256, 0, stream>>>(
            W1, DD, 0, xT, DD, sBD, b1, nullptr,
            nullptr, 0, 0, hT, HH, sTH, DD);
    // G5: rep[d,t] = sum_h W2[d,h] * hT[t,h] + b2[d] -> rep (f32, ld 1025)
    gemm_wmma_128x128<true, false, false, true, true, false, false, true, false>
        <<<dim3(TT / 128, DD / 128, BCH), 256, 0, stream>>>(
            W2, HH, 0, hT, HH, sTH, b2, nullptr,
            rep + (size_t)b0 * DD * (TT + 1), TT + 1, (long long)DD * (TT + 1),
            nullptr, 0, 0, HH);
  }

  // Ex path: mean over value[...,25:1023], then tiny MLP into rep[...,1024]
  mean_kernel  <<<dim3(BB * DD / 8),  256, 0, stream>>>(value, xex);
  exmlp1_kernel<<<dim3(BB * HH / 8),  256, 0, stream>>>(xex, W1, b1, hex);
  exmlp2_kernel<<<dim3(BB * DD / 8),  256, 0, stream>>>(hex, W2, b2, rep);
}
